// BinaryLinear_59863254172672
// MI455X (gfx1250) — 8-bit matrix operand order repaired and checked by reading, compiled, not run on hardware
//
#include <hip/hip_runtime.h>

typedef int v8i  __attribute__((ext_vector_type(8)));
typedef int v4i_ __attribute__((vector_size(16)));

#define TILE_M 128
#define TILE_N 128
#define TILE_K 64
#define THREADS 256

#if defined(__gfx1250__) && __has_builtin(__builtin_amdgcn_global_load_async_to_lds_b128) && \
    __has_builtin(__builtin_amdgcn_s_wait_asynccnt)
#define HAVE_ASYNC_LDS 1
#else
#define HAVE_ASYNC_LDS 0
#endif

__global__ __launch_bounds__(THREADS) void binarize_f32_i8(
    const float* __restrict__ in, signed char* __restrict__ out, int n4)
{
    int i = blockIdx.x * blockDim.x + threadIdx.x;
    if (i >= n4) return;
    float4 v = reinterpret_cast<const float4*>(in)[i];
    int b0 = (v.x > 0.f) ? 1 : ((v.x < 0.f) ? -1 : 0);
    int b1 = (v.y > 0.f) ? 1 : ((v.y < 0.f) ? -1 : 0);
    int b2 = (v.z > 0.f) ? 1 : ((v.z < 0.f) ? -1 : 0);
    int b3 = (v.w > 0.f) ? 1 : ((v.w < 0.f) ? -1 : 0);
    int packed = (b0 & 0xFF) | ((b1 & 0xFF) << 8) | ((b2 & 0xFF) << 16) | ((b3 & 0xFF) << 24);
    reinterpret_cast<int*>(out)[i] = packed;
}

__device__ __forceinline__ void stage_tiles(
    const signed char* __restrict__ A, const signed char* __restrict__ B,
    signed char* sA, signed char* sB,
    int tid, int K, int bm, int bn, int k0)
{
    #pragma unroll
    for (int i = 0; i < 2; ++i) {
        const int c   = tid + i * THREADS;
        const int row = c >> 2;
        const int off = (c & 3) * 16;
        const signed char* ga = A + (size_t)(bm + row) * K + k0 + off;
        const signed char* gb = B + (size_t)(bn + row) * K + k0 + off;
        signed char* la = sA + c * 16;
        signed char* lb = sB + c * 16;
#if HAVE_ASYNC_LDS
        __builtin_amdgcn_global_load_async_to_lds_b128((v4i_*)ga, (v4i_*)la, 0, 0);
        __builtin_amdgcn_global_load_async_to_lds_b128((v4i_*)gb, (v4i_*)lb, 0, 0);
#else
        *reinterpret_cast<int4*>(la) = *reinterpret_cast<const int4*>(ga);
        *reinterpret_cast<int4*>(lb) = *reinterpret_cast<const int4*>(gb);
#endif
    }
}

__device__ __forceinline__ void wait_stage()
{
#if HAVE_ASYNC_LDS
    __builtin_amdgcn_s_wait_asynccnt(0);
#endif
}

__global__ __launch_bounds__(THREADS, 2) void bgemm_wmma_iu8(
    const signed char* __restrict__ A, const signed char* __restrict__ B,
    float* __restrict__ C, int M, int N, int K)
{
    __shared__ __align__(16) signed char lA[2][TILE_M * TILE_K];
    __shared__ __align__(16) signed char lB[2][TILE_N * TILE_K];

    const int tid   = threadIdx.x;
    const int lane  = tid & 31;
    const int wave  = tid >> 5;
    const int wm    = wave >> 1;
    const int wn    = wave & 1;
    const int lhalf = lane >> 4;
    const int l15   = lane & 15;

    const int bm = blockIdx.y * TILE_M;
    const int bn = blockIdx.x * TILE_N;

    v8i acc[2][4];
    #pragma unroll
    for (int mt = 0; mt < 2; ++mt)
        #pragma unroll
        for (int nt = 0; nt < 4; ++nt)
            #pragma unroll
            for (int e = 0; e < 8; ++e)
                acc[mt][nt][e] = 0;

    stage_tiles(A, B, lA[0], lB[0], tid, K, bm, bn, 0);
    wait_stage();
    __syncthreads();

    for (int k0 = 0; k0 < K; k0 += TILE_K) {
        const int cur = (k0 / TILE_K) & 1;
        const int nxt = cur ^ 1;

        if (k0 + TILE_K < K)
            stage_tiles(A, B, lA[nxt], lB[nxt], tid, K, bm, bn, k0 + TILE_K);

        const signed char* aBuf = lA[cur];
        const signed char* bBuf = lB[cur];

        v8i afrag[2];
        #pragma unroll
        for (int mt = 0; mt < 2; ++mt) {
            const int arow = wm * 32 + mt * 16 + l15;
            const signed char* ap = aBuf + arow * TILE_K;
            #pragma unroll
            for (int p = 0; p < 4; ++p) {
                const int koff = (p >> 1) * 32 + (p & 1) * 8 + lhalf * 16;
                int2 d = *reinterpret_cast<const int2*>(ap + koff);
                afrag[mt][2 * p]     = d.x;
                afrag[mt][2 * p + 1] = d.y;
            }
        }

        v8i bfrag[4];
        #pragma unroll
        for (int nt = 0; nt < 4; ++nt) {
            const int bcol = wn * 64 + nt * 16 + l15;
            const signed char* bp = bBuf + bcol * TILE_K + lhalf * 16;
            int4 lo = *reinterpret_cast<const int4*>(bp);
            int4 hi = *reinterpret_cast<const int4*>(bp + 32);
            bfrag[nt][0] = lo.x; bfrag[nt][1] = lo.y;
            bfrag[nt][2] = lo.z; bfrag[nt][3] = lo.w;
            bfrag[nt][4] = hi.x; bfrag[nt][5] = hi.y;
            bfrag[nt][6] = hi.z; bfrag[nt][7] = hi.w;
        }

        #pragma unroll
        for (int mt = 0; mt < 2; ++mt)
            #pragma unroll
            for (int nt = 0; nt < 4; ++nt)
                acc[mt][nt] = __builtin_amdgcn_wmma_i32_16x16x64_iu8(
                     true, afrag[mt],
                     true, bfrag[nt],
                    acc[mt][nt],
                     false,  false);

        wait_stage();
        __syncthreads();
    }

    #pragma unroll
    for (int mt = 0; mt < 2; ++mt) {
        #pragma unroll
        for (int nt = 0; nt < 4; ++nt) {
            const int col = bn + wn * 64 + nt * 16 + l15;
            #pragma unroll
            for (int v = 0; v < 8; ++v) {
                const int row = bm + wm * 32 + mt * 16 + lhalf * 8 + v;
                C[(size_t)row * N + col] = (float)acc[mt][nt][v];
            }
        }
    }
}

extern "C" void kernel_launch(void* const* d_in, const int* in_sizes, int n_in,
                              void* d_out, int out_size, void* d_ws, size_t ws_size,
                              hipStream_t stream)
{
    const float* x = (const float*)d_in[0];
    const float* w = (const float*)d_in[1];
    float* out = (float*)d_out;

    const int K = 4096;
    const int M = in_sizes[0] / K;
    const int N = in_sizes[1] / K;

    signed char* xb = (signed char*)d_ws;
    signed char* wb = xb + (size_t)M * K;

    int nx4 = (M * K) / 4;
    int nw4 = (N * K) / 4;
    binarize_f32_i8<<<(nx4 + THREADS - 1) / THREADS, THREADS, 0, stream>>>(x, xb, nx4);
    binarize_f32_i8<<<(nw4 + THREADS - 1) / THREADS, THREADS, 0, stream>>>(w, wb, nw4);

    dim3 grid(N / TILE_N, M / TILE_M);
    bgemm_wmma_iu8<<<grid, THREADS, 0, stream>>>(xb, wb, out, M, N, K);
}
